// GDNBlockWrapper_64957085385366
// MI455X (gfx1250) — matrix operand order repaired and checked by reading, not compiled, not run on hardware
//
#include <hip/hip_runtime.h>
#include <math.h>

// ---------------- types ----------------
typedef __attribute__((ext_vector_type(16))) _Float16 hv16;
typedef __attribute__((ext_vector_type(8)))  _Float16 hv8;
typedef __attribute__((ext_vector_type(4)))  _Float16 hv4;
typedef __attribute__((ext_vector_type(8)))  float    fv8;
typedef __attribute__((ext_vector_type(4)))  float    fv4;

// ---------------- problem constants ----------------
constexpr int cB  = 2;
constexpr int cT  = 1024;
constexpr int cD  = 1024;
constexpr int cH  = 8;
constexpr int cDK = 128;
constexpr int cDV = 128;
constexpr int cKW = 4;
constexpr int cF  = 4096;
constexpr int cL  = 3;
constexpr int cKD = cH * cDK;           // 1024
constexpr int cVD = cH * cDV;           // 1024
constexpr int cCD = 2 * cKD + cVD;      // 3072
constexpr int cM  = cB * cT;            // 2048 rows

static __device__ __forceinline__ float dev_sigmoid(float x) {
  return 1.0f / (1.0f + expf(-x));
}

static __device__ __forceinline__ float waveSum(float x) {
#pragma unroll
  for (int o = 16; o > 0; o >>= 1) x += __shfl_xor(x, o, 32);
  return x;
}

// ---------------- RMSNorm (+optional mask) ----------------
// rows = cM, block 256, each thread handles 4 of D=1024 elements
__global__ __launch_bounds__(256) void rmsnorm_kernel(
    const float* __restrict__ x, const float* __restrict__ w,
    const float* __restrict__ mask, float* __restrict__ y) {
  const int row = blockIdx.x;
  const int tid = threadIdx.x;
  const float* xr = x + (size_t)row * cD;
  float v0 = xr[tid], v1 = xr[tid + 256], v2 = xr[tid + 512], v3 = xr[tid + 768];
  float s = v0 * v0 + v1 * v1 + v2 * v2 + v3 * v3;
  __shared__ float red[8];
  s = waveSum(s);
  if ((tid & 31) == 0) red[tid >> 5] = s;
  __syncthreads();
  if (tid == 0) {
    float t = 0.0f;
#pragma unroll
    for (int i = 0; i < 8; ++i) t += red[i];
    red[0] = t;
  }
  __syncthreads();
  const float scale = rsqrtf(red[0] * (1.0f / cD) + 1e-6f);
  const float m = mask ? mask[row] : 1.0f;
  float* yr = y + (size_t)row * cD;
  yr[tid]       = v0 * scale * w[tid]       * m;
  yr[tid + 256] = v1 * scale * w[tid + 256] * m;
  yr[tid + 512] = v2 * scale * w[tid + 512] * m;
  yr[tid + 768] = v3 * scale * w[tid + 768] * m;
}

// ---------------- WMMA f16 GEMM: C[M,N] (+)= A[M,K] * B[K,N] ----------------
// Tile 64x64x32, 128 threads = 4 waves, each wave computes a 32x32 subtile
// (2x2 grid of v_wmma_f32_16x16x32_f16). M%64==0, N%64==0, K%32==0 assumed.
// ACC is a compile-time flag so the epilogue is branch-free. Prefetches of the
// next K tile are unconditional: gfx1250 DEV-scope prefetch is speculative
// (silently dropped on translation failure), so running one tile past the end
// is safe and avoids per-load branches in the hot loop.
template <bool ACC>
__global__ __launch_bounds__(128) void gemm_wmma_kernel(
    const float* __restrict__ A, const float* __restrict__ Bm,
    float* __restrict__ C, int M, int N, int K) {
  __shared__ _Float16 As[64][40];   // [row][k], pad to 40 halves (80B rows)
  __shared__ _Float16 Bs[64][40];   // [col][k], transposed B tile

  const int tid  = threadIdx.x;
  const int lane = tid & 31;
  const int wave = tid >> 5;
  const int wr = (wave >> 1) * 32;  // wave M origin within tile
  const int wc = (wave & 1) * 32;   // wave N origin within tile
  const int bn0 = blockIdx.x * 64;
  const int bm0 = blockIdx.y * 64;

  fv8 acc[2][2];
#pragma unroll
  for (int i = 0; i < 2; ++i)
#pragma unroll
    for (int j = 0; j < 2; ++j)
#pragma unroll
      for (int r = 0; r < 8; ++r) acc[i][j][r] = 0.0f;

  for (int k0 = 0; k0 < K; k0 += 32) {
    // stage A tile 64x32 (f32 -> f16)
#pragma unroll
    for (int it = 0; it < 4; ++it) {
      int idx = tid + it * 128;       // 0..511 float4 slots
      int r = idx >> 3;               // row 0..63
      int c = (idx & 7) << 2;         // k 0,4,..,28
      const float* gp = A + (size_t)(bm0 + r) * K + k0 + c;
      fv4 av = *(const fv4*)gp;
      __builtin_prefetch(gp + 32, 0, 1);  // next K tile (speculative)
      hv4 hv = { (_Float16)av.x, (_Float16)av.y, (_Float16)av.z, (_Float16)av.w };
      *(hv4*)(&As[r][c]) = hv;
    }
    // stage B tile 32x64 transposed into [col][k]
#pragma unroll
    for (int it = 0; it < 4; ++it) {
      int idx = tid + it * 128;
      int kr = idx >> 4;              // k row 0..31
      int n4 = (idx & 15) << 2;       // col group
      const float* gp = Bm + (size_t)(k0 + kr) * N + bn0 + n4;
      fv4 bv = *(const fv4*)gp;
      __builtin_prefetch(gp + (size_t)32 * N, 0, 1);
      Bs[n4 + 0][kr] = (_Float16)bv.x;
      Bs[n4 + 1][kr] = (_Float16)bv.y;
      Bs[n4 + 2][kr] = (_Float16)bv.z;
      Bs[n4 + 3][kr] = (_Float16)bv.w;
    }
    __syncthreads();

    const int lrow = lane & 15;
    const int hi16 = lane >> 4;
    hv16 afr[2], bfr[2];
    // A frag: lane<16 holds K 0-15; lane>=16 holds K 16-31 (contiguous)
    {
      const int kb = hi16 * 16;
#pragma unroll
      for (int i = 0; i < 2; ++i) {
        const _Float16* p = &As[wr + i * 16 + lrow][kb];
        hv8 lo = *(const hv8*)p;
        hv8 hi = *(const hv8*)(p + 8);
        afr[i] = __builtin_shufflevector(lo, hi, 0, 1, 2, 3, 4, 5, 6, 7, 8, 9,
                                         10, 11, 12, 13, 14, 15);
      }
    }
    // B frag: lane<16 holds K 0-15; lane>=16 holds K 16-31 (contiguous)
    {
      const int kb = hi16 * 16;
#pragma unroll
      for (int j = 0; j < 2; ++j) {
        const _Float16* p = &Bs[wc + j * 16 + lrow][kb];
        hv8 lo = *(const hv8*)p;
        hv8 hi = *(const hv8*)(p + 8);
        bfr[j] = __builtin_shufflevector(lo, hi, 0, 1, 2, 3, 4, 5, 6, 7, 8, 9,
                                         10, 11, 12, 13, 14, 15);
      }
    }
#pragma unroll
    for (int i = 0; i < 2; ++i)
#pragma unroll
      for (int j = 0; j < 2; ++j)
        acc[i][j] = __builtin_amdgcn_wmma_f32_16x16x32_f16(
            false, afr[i], false, bfr[j], (short)0, acc[i][j], false, false);
    __syncthreads();
  }

  // C layout: VGPR r holds row (r + 8*(lane>=16)), col = lane&15
  const int crow = (lane >> 4) * 8;
  const int ccol = lane & 15;
#pragma unroll
  for (int i = 0; i < 2; ++i)
#pragma unroll
    for (int j = 0; j < 2; ++j) {
      const int gr = bm0 + wr + i * 16 + crow;
      const int gc = bn0 + wc + j * 16 + ccol;
#pragma unroll
      for (int r = 0; r < 8; ++r) {
        size_t off = (size_t)(gr + r) * N + gc;
        if (ACC) {
          C[off] += acc[i][j][r];
        } else {
          C[off] = acc[i][j][r];
        }
      }
    }
}

// ---------------- cached causal depthwise conv1d + SiLU (in-place) --------
// one thread per (b, channel); writes new conv state directly to d_out
__global__ __launch_bounds__(256) void conv_silu_kernel(
    const float* __restrict__ cw,      // [CD,KW] layer slice
    const float* __restrict__ cstate,  // [B,CD,KW]
    float* __restrict__ xio,           // [B,T,CD] in/out
    float* __restrict__ ncs) {         // [B,CD,KW] out
  const int gid = blockIdx.x * 256 + threadIdx.x;
  if (gid >= cB * cCD) return;
  const int b = gid / cCD, c = gid % cCD;
  const float w0 = cw[c * cKW + 0], w1 = cw[c * cKW + 1];
  const float w2 = cw[c * cKW + 2], w3 = cw[c * cKW + 3];
  const float* st = cstate + ((size_t)b * cCD + c) * cKW;
  float f0 = st[0], f1 = st[1], f2 = st[2], f3 = st[3];
  float* col = xio + (size_t)b * cT * cCD + c;
  for (int t = 0; t < cT; ++t) {
    const float xt = col[(size_t)t * cCD];
    const float y = w0 * f1 + w1 * f2 + w2 * f3 + w3 * xt;
    col[(size_t)t * cCD] = y * dev_sigmoid(y);  // silu
    f0 = f1; f1 = f2; f2 = f3; f3 = xt;
  }
  float* np = ncs + ((size_t)b * cCD + c) * cKW;
  np[0] = f0; np[1] = f1; np[2] = f2; np[3] = f3;
}

// ---------------- beta = sigmoid(x@Wb), decay = exp(-exp(A)*softplus(x@Wa+dtb))
// one wave per (b,t,h); block = 8 waves
__global__ __launch_bounds__(256) void beta_decay_kernel(
    const float* __restrict__ x, const float* __restrict__ Wb,
    const float* __restrict__ Wa, const float* __restrict__ dtb,
    const float* __restrict__ Alog, float* __restrict__ beta,
    float* __restrict__ decay) {
  const int wave = threadIdx.x >> 5, lane = threadIdx.x & 31;
  const int idx = blockIdx.x * 8 + wave;  // 0..cM*cH-1
  const int h = idx % cH, bt = idx / cH;
  const float* xr = x + (size_t)bt * cD;
  float sb = 0.0f, sa = 0.0f;
  for (int d = lane; d < cD; d += 32) {
    const float xv = xr[d];
    sb += xv * Wb[d * cH + h];
    sa += xv * Wa[d * cH + h];
  }
  sb = waveSum(sb);
  sa = waveSum(sa);
  if (lane == 0) {
    beta[idx] = dev_sigmoid(sb);
    const float z = sa + dtb[h];
    const float sp = (z > 20.0f) ? z : log1pf(expf(z));
    decay[idx] = expf(-expf(Alog[h]) * sp);
  }
}

// ---------------- per-head L2 normalize q (and *DK^-0.5) and k, in-place ----
__global__ __launch_bounds__(128) void qk_l2norm_kernel(float* __restrict__ qkv) {
  const int bh = blockIdx.x;  // bt*H + h
  const int h = bh % cH, bt = bh / cH;
  const int d = threadIdx.x;
  float* qp = qkv + (size_t)bt * cCD + h * cDK;
  float* kp = qkv + (size_t)bt * cCD + cKD + h * cDK;
  const float qv = qp[d], kv = kp[d];
  __shared__ float rq[4], rk[4];
  const float sq = waveSum(qv * qv);
  const float sk = waveSum(kv * kv);
  if ((d & 31) == 0) { rq[d >> 5] = sq; rk[d >> 5] = sk; }
  __syncthreads();
  const float tq = rq[0] + rq[1] + rq[2] + rq[3];
  const float tk = rk[0] + rk[1] + rk[2] + rk[3];
  qp[d] = qv * rsqrtf(tq + 1e-6f) * 0.08838834764831845f;  // DK^-0.5
  kp[d] = kv * rsqrtf(tk + 1e-6f);
}

// ---------------- gated delta rule scan ----------------
// grid = B*H; block 256. Thread (tid) owns S[k][v] for v = tid&127,
// k in [(tid>>7)*64, +64). State kept in registers (64 f32/thread).
__global__ __launch_bounds__(256) void gdn_scan_kernel(
    const float* __restrict__ qkv, const float* __restrict__ beta,
    const float* __restrict__ decay, const float* __restrict__ S0,
    float* __restrict__ o, float* __restrict__ Sout) {
  const int bh = blockIdx.x;
  const int b = bh / cH, h = bh % cH;
  const int tid = threadIdx.x;
  const int v = tid & 127;
  const int kr0 = (tid >> 7) * 64;

  float S[64];
  const float* S0p = S0 + (size_t)bh * cDK * cDV;
#pragma unroll
  for (int j = 0; j < 64; ++j) S[j] = S0p[(size_t)(kr0 + j) * cDV + v];

  __shared__ float qs[128], ks[128], vs[128], dvs[128], red[256];

  for (int t = 0; t < cT; ++t) {
    const size_t base = ((size_t)(b * cT + t)) * cCD;
    if (tid < 128) {
      qs[tid] = qkv[base + h * cDK + tid];
      ks[tid] = qkv[base + cKD + h * cDK + tid];
      vs[tid] = qkv[base + 2 * cKD + h * cDV + tid];
    }
    const size_t gidx = (size_t)(b * cT + t) * cH + h;
    const float g = decay[gidx];
    const float bet = beta[gidx];
    __syncthreads();

    float pred = 0.0f;
#pragma unroll
    for (int j = 0; j < 64; ++j) {  // S *= g; pred = k . S
      S[j] *= g;
      pred += ks[kr0 + j] * S[j];
    }
    red[tid] = pred;
    __syncthreads();
    if (tid < 128) dvs[tid] = bet * (vs[tid] - (red[tid] + red[tid + 128]));
    __syncthreads();

    const float d = dvs[v];
    float out = 0.0f;
#pragma unroll
    for (int j = 0; j < 64; ++j) {  // S += k (x) dv; out = q . S
      S[j] += ks[kr0 + j] * d;
      out += qs[kr0 + j] * S[j];
    }
    red[tid] = out;
    __syncthreads();
    if (tid < 128)
      o[((size_t)(b * cT + t)) * cVD + h * cDV + tid] = red[tid] + red[tid + 128];
    __syncthreads();
  }

  float* Sp = Sout + (size_t)bh * cDK * cDV;
#pragma unroll
  for (int j = 0; j < 64; ++j) Sp[(size_t)(kr0 + j) * cDV + v] = S[j];
}

// ---------------- gated RMSNorm * silu(gate), in-place on o ----------------
__global__ __launch_bounds__(128) void gated_norm_kernel(
    float* __restrict__ o, const float* __restrict__ g,
    const float* __restrict__ gnw) {
  const int bh = blockIdx.x;
  const int h = bh % cH, bt = bh / cH;
  const int d = threadIdx.x;
  const size_t base = (size_t)bt * cVD + h * cDV;
  const float ov = o[base + d];
  __shared__ float r[4];
  const float s = waveSum(ov * ov);
  if ((d & 31) == 0) r[d >> 5] = s;
  __syncthreads();
  const float mean = (r[0] + r[1] + r[2] + r[3]) * (1.0f / cDV);
  const float gv = g[base + d];
  o[base + d] = ov * rsqrtf(mean + 1e-6f) * gnw[d] * (gv * dev_sigmoid(gv));
}

// ---------------- SwiGLU elementwise: gt = silu(gt) * up ----------------
__global__ __launch_bounds__(256) void swiglu_kernel(
    float* __restrict__ gt, const float* __restrict__ up, int n) {
  const int i = blockIdx.x * 256 + threadIdx.x;
  if (i < n) {
    const float g = gt[i];
    gt[i] = g * dev_sigmoid(g) * up[i];
  }
}

__global__ __launch_bounds__(256) void copy_kernel(
    const float* __restrict__ src, float* __restrict__ dst, int n) {
  const int i = blockIdx.x * 256 + threadIdx.x;
  if (i < n) dst[i] = src[i];
}

// ---------------- host launcher ----------------
extern "C" void kernel_launch(void* const* d_in, const int* in_sizes, int n_in,
                              void* d_out, int out_size, void* d_ws,
                              size_t ws_size, hipStream_t stream) {
  const float* hidden = (const float*)d_in[0];
  const float* mask   = (const float*)d_in[1];
  const float* Wqkv   = (const float*)d_in[8];
  const float* convw  = (const float*)d_in[9];
  const float* Wb     = (const float*)d_in[10];
  const float* Wa     = (const float*)d_in[11];
  const float* Wg     = (const float*)d_in[12];
  const float* Wo     = (const float*)d_in[13];
  const float* Wgate  = (const float*)d_in[14];
  const float* Wup    = (const float*)d_in[15];
  const float* Wdown  = (const float*)d_in[16];
  const float* dtb    = (const float*)d_in[17];
  const float* Alog   = (const float*)d_in[18];
  const float* gnw    = (const float*)d_in[19];
  const float* ln1    = (const float*)d_in[20];
  const float* ln2    = (const float*)d_in[21];

  float* out = (float*)d_out;
  float* ws  = (float*)d_ws;

  const size_t NH = (size_t)cM * cD;  // 2,097,152 floats
  float* hbuf  = ws;
  float* xnorm = hbuf + NH;
  float* qkv   = xnorm + NH;
  float* obuf  = qkv + (size_t)cM * cCD;
  float* gbuf  = obuf + NH;
  float* gtbuf = gbuf + NH;
  float* upbuf = gtbuf + (size_t)cM * cF;
  float* betab = upbuf + (size_t)cM * cF;
  float* decb  = betab + (size_t)cM * cH;

  const size_t NCS = (size_t)cB * cCD * cKW;        // 24576
  const size_t NRS = (size_t)cB * cH * cDK * cDV;   // 262144

  copy_kernel<<<(unsigned)(NH / 256), 256, 0, stream>>>(hidden, hbuf, (int)NH);

  for (int l = 0; l < cL; ++l) {
    const float* cs_in = (const float*)d_in[2 + 2 * l];
    const float* rs_in = (const float*)d_in[3 + 2 * l];
    float* ncs = out + NH + (size_t)l * (NCS + NRS);
    float* nrs = ncs + NCS;

    // x = rms(h, ln1) * mask
    rmsnorm_kernel<<<cM, 256, 0, stream>>>(hbuf, ln1 + (size_t)l * cD, mask, xnorm);
    // qkv = x @ Wqkv
    gemm_wmma_kernel<false><<<dim3(cCD / 64, cM / 64), 128, 0, stream>>>(
        xnorm, Wqkv + (size_t)l * cD * cCD, qkv, cM, cCD, cD);
    // causal depthwise conv + silu (in place), new conv state -> d_out
    conv_silu_kernel<<<(cB * cCD) / 256, 256, 0, stream>>>(
        convw + (size_t)l * cCD * cKW, cs_in, qkv, ncs);
    // beta / decay from x
    beta_decay_kernel<<<(cM * cH) / 8, 256, 0, stream>>>(
        xnorm, Wb + (size_t)l * cD * cH, Wa + (size_t)l * cD * cH,
        dtb + (size_t)l * cH, Alog + (size_t)l * cH, betab, decb);
    // l2-normalize q (scaled) and k in place
    qk_l2norm_kernel<<<cM * cH, 128, 0, stream>>>(qkv);
    // gated delta rule scan; new recurrent state -> d_out
    gdn_scan_kernel<<<cB * cH, 256, 0, stream>>>(qkv, betab, decb, rs_in, obuf, nrs);
    // g = x @ Wg
    gemm_wmma_kernel<false><<<dim3(cVD / 64, cM / 64), 128, 0, stream>>>(
        xnorm, Wg + (size_t)l * cD * cVD, gbuf, cM, cVD, cD);
    // o = rms(o, gnw) * silu(g) in place
    gated_norm_kernel<<<cM * cH, 128, 0, stream>>>(obuf, gbuf, gnw + (size_t)l * cDV);
    // h += o @ Wo
    gemm_wmma_kernel<true><<<dim3(cD / 64, cM / 64), 128, 0, stream>>>(
        obuf, Wo + (size_t)l * cVD * cD, hbuf, cM, cD, cVD);
    // y = rms(h, ln2)  (reuse xnorm)
    rmsnorm_kernel<<<cM, 256, 0, stream>>>(hbuf, ln2 + (size_t)l * cD, nullptr, xnorm);
    // gate / up projections
    gemm_wmma_kernel<false><<<dim3(cF / 64, cM / 64), 128, 0, stream>>>(
        xnorm, Wgate + (size_t)l * cD * cF, gtbuf, cM, cF, cD);
    gemm_wmma_kernel<false><<<dim3(cF / 64, cM / 64), 128, 0, stream>>>(
        xnorm, Wup + (size_t)l * cD * cF, upbuf, cM, cF, cD);
    // gt = silu(gt) * up
    swiglu_kernel<<<(cM * cF) / 256, 256, 0, stream>>>(gtbuf, upbuf, cM * cF);
    // h += gt @ Wdown
    gemm_wmma_kernel<true><<<dim3(cD / 64, cM / 64), 128, 0, stream>>>(
        gtbuf, Wdown + (size_t)l * cF * cD, hbuf, cM, cD, cF);
  }

  copy_kernel<<<(unsigned)(NH / 256), 256, 0, stream>>>(hbuf, out, (int)NH);

  (void)in_sizes; (void)n_in; (void)out_size; (void)ws_size;
}
